// LinearAttention_17162689315391
// MI455X (gfx1250) — compile-verified, matrix operand order checked by reading, not run on hardware
//
#include <hip/hip_runtime.h>
#include <stdint.h>

// ---------------------------------------------------------------------------
// CDNA5 (gfx1250) wave32 WMMA implementation of the normalized-attention block
// ---------------------------------------------------------------------------

typedef __bf16 bf16_t;
typedef __attribute__((ext_vector_type(16))) __bf16 v16bf;
typedef __attribute__((ext_vector_type(8)))  float  v8f;

constexpr int kHeads = 8;
constexpr int kDh    = 64;
constexpr int kDim   = 512;
constexpr int kNtok  = 4096;
constexpr int kB     = 8;
constexpr int kToks  = kB * kNtok;   // 32768 tokens
constexpr int kQkvF  = 1536;         // 3 * heads * dh
constexpr int kSeq   = kToks;        // columns of feature-major qkvT

// Load one WMMA 16-bit operand row chunk (K=32) honoring the CDNA5 lane/K
// interleave: lanes 0-15 hold K {k0..k0+7, k0+16..k0+23}, lanes 16-31 hold
// K {k0+8..k0+15, k0+24..k0+31}.  rowp must be 16B-aligned per 8-element step.
static __device__ __forceinline__ v16bf ld_k32(const bf16_t* rowp, int k0, int sel) {
  union { uint4 u[2]; v16bf v; } cvt;
  cvt.u[0] = *reinterpret_cast<const uint4*>(rowp + k0 + sel * 8);
  cvt.u[1] = *reinterpret_cast<const uint4*>(rowp + k0 + 16 + sel * 8);
  return cvt.v;
}

static __device__ __forceinline__ v8f wmma_bf16(v16bf a, v16bf b, v8f c) {
  // (neg_a, A, neg_b, B, c_mod, C, reuse_a, reuse_b)
  return __builtin_amdgcn_wmma_f32_16x16x32_bf16(false, a, false, b, (short)0, c,
                                                 false, false);
}

static __device__ __forceinline__ unsigned pack2bf(float lo, float hi) {
  union { __bf16 b[2]; unsigned u; } p;
  p.b[0] = (__bf16)lo; p.b[1] = (__bf16)hi;
  return p.u;
}

// ---------------------------------------------------------------------------
// K0: transpose + downconvert weights once (tiny): wqkvT[1536,512], woutT[512,512]
// ---------------------------------------------------------------------------
__global__ void prep_weights_kernel(const float* __restrict__ wqkv,
                                    const float* __restrict__ wout,
                                    bf16_t* __restrict__ wqkvT,
                                    bf16_t* __restrict__ woutT) {
  int idx = blockIdx.x * blockDim.x + threadIdx.x;
  if (idx < kQkvF * kDim) {
    int f = idx / kDim, k = idx % kDim;           // wqkvT[f][k] = wqkv[k][f]
    wqkvT[idx] = (bf16_t)wqkv[(size_t)k * kQkvF + f];
  } else {
    int idx2 = idx - kQkvF * kDim;
    if (idx2 < kDim * kDim) {
      int o = idx2 / kDim, i = idx2 % kDim;       // woutT[o][i] = wout[i][o]
      woutT[idx2] = (bf16_t)wout[(size_t)i * kDim + o];
    }
  }
}

// ---------------------------------------------------------------------------
// K1: RMSNorm per token row -> bf16 xn [32768, 512]  (one wave per row)
// ---------------------------------------------------------------------------
__global__ void rmsnorm_kernel(const float* __restrict__ x,
                               const float* __restrict__ gamma,
                               bf16_t* __restrict__ xn) {
  int wave = threadIdx.x >> 5, lane = threadIdx.x & 31;
  int row  = blockIdx.x * 8 + wave;               // 0..32767
  const float* rp = x + (size_t)row * kDim;
  float ss = 0.f;
  #pragma unroll
  for (int i = lane; i < kDim; i += 32) { float v = rp[i]; ss += v * v; }
  #pragma unroll
  for (int off = 16; off; off >>= 1) ss += __shfl_xor(ss, off, 32);
  float s = 22.62741699796952f / fmaxf(sqrtf(ss), 1e-12f);   // sqrt(512)/||x||
  bf16_t* op = xn + (size_t)row * kDim;
  #pragma unroll
  for (int i = lane; i < kDim; i += 32) op[i] = (bf16_t)(rp[i] * s * gamma[i]);
}

// ---------------------------------------------------------------------------
// K2: qkvT[1536, 32768] = wqkvT[1536,512] @ xn^T   (feature-major output, bf16)
// Register-blocked: each wave computes a 64(M=f) x 32(N=tok) macro-tile
// (4x2 WMMA tiles, 8 accumulators), reusing each operand load across tiles.
// ---------------------------------------------------------------------------
__global__ void qkv_gemm_kernel(const bf16_t* __restrict__ wqkvT,
                                const bf16_t* __restrict__ xn,
                                bf16_t* __restrict__ qkvT) {
  int wave = threadIdx.x >> 5, lane = threadIdx.x & 31;
  int l15 = lane & 15, sel = lane >> 4;
  int t  = blockIdx.x * 8 + wave;                 // 0..24575
  int tm = t / (kToks / 32);                      // feature block 0..23 (64 rows)
  int tn = t % (kToks / 32);                      // token block 0..1023 (32 cols)
  const bf16_t* arow[4];
  const bf16_t* brow[2];
  #pragma unroll
  for (int i = 0; i < 4; ++i)
    arow[i] = wqkvT + (size_t)(tm * 64 + i * 16 + l15) * kDim;
  #pragma unroll
  for (int j = 0; j < 2; ++j)
    brow[j] = xn + (size_t)(tn * 32 + j * 16 + l15) * kDim;

  v8f acc[4][2] = {};
  for (int k0 = 0; k0 < kDim; k0 += 32) {
    v16bf a[4], b[2];
    #pragma unroll
    for (int i = 0; i < 4; ++i) a[i] = ld_k32(arow[i], k0, sel);
    #pragma unroll
    for (int j = 0; j < 2; ++j) b[j] = ld_k32(brow[j], k0, sel);
    #pragma unroll
    for (int i = 0; i < 4; ++i)
      #pragma unroll
      for (int j = 0; j < 2; ++j)
        acc[i][j] = wmma_bf16(a[i], b[j], acc[i][j]);
  }
  #pragma unroll
  for (int i = 0; i < 4; ++i) {
    int fbase = tm * 64 + i * 16 + sel * 8;       // C layout: VGPR r -> M = r (+8 hi lanes)
    #pragma unroll
    for (int j = 0; j < 2; ++j) {
      int tok = tn * 32 + j * 16 + l15;
      #pragma unroll
      for (int r = 0; r < 8; ++r)
        qkvT[(size_t)(fbase + r) * kSeq + tok] = (bf16_t)acc[i][j][r];
    }
  }
}

// ---------------------------------------------------------------------------
// K3: inverse L2 norms over n for q (rows 0..511) and k (rows 512..1023),
// per batch segment of 4096 columns.  invn[b*1024 + f].
// ---------------------------------------------------------------------------
__global__ void rownorm_kernel(const bf16_t* __restrict__ qkvT,
                               float* __restrict__ invn) {
  int wave = threadIdx.x >> 5, lane = threadIdx.x & 31;
  int w = blockIdx.x * 8 + wave;                  // 0..8191
  int f = w >> 3, b = w & 7;
  const bf16_t* rp = qkvT + (size_t)f * kSeq + b * kNtok;
  float ss = 0.f;
  for (int i = lane; i < kNtok; i += 32) { float v = (float)rp[i]; ss += v * v; }
  #pragma unroll
  for (int off = 16; off; off >>= 1) ss += __shfl_xor(ss, off, 32);
  if (lane == 0) invn[b * 1024 + f] = 1.f / fmaxf(sqrtf(ss), 1e-12f);
}

// ---------------------------------------------------------------------------
// K4: per (b,h): raw sim = q[64,4096] @ k^T  via WMMA (K = n = 4096), then
// apply separable normalization * 8 * exp(temperature) and softmax over e.
// Output attn bf16 [bh][64][64].  One workgroup (8 waves) per (b,h).
// ---------------------------------------------------------------------------
__global__ void sim_softmax_kernel(const bf16_t* __restrict__ qkvT,
                                   const float* __restrict__ invn,
                                   const float* __restrict__ temperature,
                                   bf16_t* __restrict__ attn) {
  __shared__ float sim[64 * 65];
  int bh = blockIdx.x, b = bh >> 3, h = bh & 7;
  int wave = threadIdx.x >> 5, lane = threadIdx.x & 31;
  int l15 = lane & 15, sel = lane >> 4;
  int tm  = (wave * 2) >> 2;                      // d-tile (shared by the wave's 2 tiles)
  int tn0 = (wave * 2) & 3;                       // e-tiles tn0, tn0+1
  const bf16_t* arow  = qkvT + (size_t)(h * kDh + tm * 16 + l15) * kSeq + b * kNtok;
  const bf16_t* brow0 = qkvT + (size_t)(512 + h * kDh + tn0 * 16 + l15) * kSeq + b * kNtok;
  const bf16_t* brow1 = brow0 + (size_t)16 * kSeq;
  v8f acc0 = {}, acc1 = {};
  for (int k0 = 0; k0 < kNtok; k0 += 32) {
    v16bf a = ld_k32(arow, k0, sel);              // A reused across both tiles
    acc0 = wmma_bf16(a, ld_k32(brow0, k0, sel), acc0);
    acc1 = wmma_bf16(a, ld_k32(brow1, k0, sel), acc1);
  }
  #pragma unroll
  for (int j = 0; j < 8; ++j) {
    int r = tm * 16 + sel * 8 + j;
    sim[r * 65 + tn0 * 16 + l15]       = acc0[j];
    sim[r * 65 + (tn0 + 1) * 16 + l15] = acc1[j];
  }
  __syncthreads();
  // Softmax over e: 4 threads per row (same wave), 16 cols each.
  int row = threadIdx.x >> 2, cs = (threadIdx.x & 3) * 16;
  float qf = invn[b * 1024 + h * kDh + row] * 8.f * __expf(temperature[h]);
  const float* kf = &invn[b * 1024 + 512 + h * kDh];
  float v[16], mx = -1e30f;
  #pragma unroll
  for (int i = 0; i < 16; ++i) {
    v[i] = sim[row * 65 + cs + i] * qf * kf[cs + i];
    mx = fmaxf(mx, v[i]);
  }
  mx = fmaxf(mx, __shfl_xor(mx, 1, 32));
  mx = fmaxf(mx, __shfl_xor(mx, 2, 32));
  float sum = 0.f;
  #pragma unroll
  for (int i = 0; i < 16; ++i) { v[i] = __expf(v[i] - mx); sum += v[i]; }
  sum += __shfl_xor(sum, 1, 32);
  sum += __shfl_xor(sum, 2, 32);
  float inv = 1.f / sum;
  bf16_t* op = attn + (size_t)bh * 4096 + row * 64 + cs;
  #pragma unroll
  for (int i = 0; i < 16; ++i) op[i] = (bf16_t)(v[i] * inv);
}

// ---------------------------------------------------------------------------
// K5: out = attn[64,64] @ v[64, n-chunk]; v tile transposed through LDS so the
// B operand is contiguous.  Output written transposed -> AO[b, n, h*64+d] bf16
// (token-major) so the final projection is a plain row-major GEMM.
// Grid: (b*h) x (4096/64) chunks; one workgroup per chunk.
// ---------------------------------------------------------------------------
__global__ void attn_v_kernel(const bf16_t* __restrict__ qkvT,
                              const bf16_t* __restrict__ attn,
                              bf16_t* __restrict__ AO) {
  __shared__ __align__(16) bf16_t vT[64 * 72];    // [n][e], padded rows
  int bh = blockIdx.x >> 6, nc = blockIdx.x & 63;
  int b = bh >> 3, h = bh & 7;
  int n0 = nc * 64;
  // Stage v[64e x 64n] into LDS transposed (coalesced global reads).
  int e = threadIdx.x >> 2, c0 = (threadIdx.x & 3) * 16;
  const bf16_t* vp = qkvT + (size_t)(1024 + h * kDh + e) * kSeq + b * kNtok + n0 + c0;
  #pragma unroll
  for (int i = 0; i < 16; ++i) vT[(c0 + i) * 72 + e] = vp[i];
  __syncthreads();
  int wave = threadIdx.x >> 5, lane = threadIdx.x & 31;
  int l15 = lane & 15, sel = lane >> 4;
  int tm = (wave * 2) >> 2, tn0 = (wave * 2) & 3;
  const bf16_t* arow  = attn + (size_t)bh * 4096 + (tm * 16 + l15) * 64;
  const bf16_t* brow0 = &vT[(tn0 * 16 + l15) * 72];
  const bf16_t* brow1 = &vT[((tn0 + 1) * 16 + l15) * 72];
  v8f acc0 = {}, acc1 = {};
  #pragma unroll
  for (int k0 = 0; k0 < 64; k0 += 32) {
    v16bf a = ld_k32(arow, k0, sel);
    acc0 = wmma_bf16(a, ld_k32(brow0, k0, sel), acc0);
    acc1 = wmma_bf16(a, ld_k32(brow1, k0, sel), acc1);
  }
  int dbase = tm * 16 + sel * 8;                  // 8 consecutive d per lane
  #pragma unroll
  for (int it = 0; it < 2; ++it) {
    v8f& acc = it ? acc1 : acc0;
    int n = n0 + (tn0 + it) * 16 + l15;
    uint4 pk;
    pk.x = pack2bf(acc[0], acc[1]);
    pk.y = pack2bf(acc[2], acc[3]);
    pk.z = pack2bf(acc[4], acc[5]);
    pk.w = pack2bf(acc[6], acc[7]);
    *reinterpret_cast<uint4*>(AO + (size_t)(b * kNtok + n) * kDim + h * kDh + dbase) = pk;
  }
}

// ---------------------------------------------------------------------------
// K6: d_out[32768, 512] f32 = AO[32768,512]bf16 @ w_out (via woutT rows for B)
// Register-blocked 64(M=token) x 32(N=outfeat) per wave (4x2 tiles).
// ---------------------------------------------------------------------------
__global__ void out_gemm_kernel(const bf16_t* __restrict__ AO,
                                const bf16_t* __restrict__ woutT,
                                float* __restrict__ out) {
  int wave = threadIdx.x >> 5, lane = threadIdx.x & 31;
  int l15 = lane & 15, sel = lane >> 4;
  int t  = blockIdx.x * 8 + wave;                 // 0..8191
  int tm = t >> 4;                                // token block 0..511 (64 rows)
  int tn = t & 15;                                // outfeat block 0..15 (32 cols)
  const bf16_t* arow[4];
  const bf16_t* brow[2];
  #pragma unroll
  for (int i = 0; i < 4; ++i)
    arow[i] = AO + (size_t)(tm * 64 + i * 16 + l15) * kDim;
  #pragma unroll
  for (int j = 0; j < 2; ++j)
    brow[j] = woutT + (size_t)(tn * 32 + j * 16 + l15) * kDim;

  v8f acc[4][2] = {};
  for (int k0 = 0; k0 < kDim; k0 += 32) {
    v16bf a[4], b[2];
    #pragma unroll
    for (int i = 0; i < 4; ++i) a[i] = ld_k32(arow[i], k0, sel);
    #pragma unroll
    for (int j = 0; j < 2; ++j) b[j] = ld_k32(brow[j], k0, sel);
    #pragma unroll
    for (int i = 0; i < 4; ++i)
      #pragma unroll
      for (int j = 0; j < 2; ++j)
        acc[i][j] = wmma_bf16(a[i], b[j], acc[i][j]);
  }
  #pragma unroll
  for (int i = 0; i < 4; ++i) {
    int tok = tm * 64 + i * 16 + sel * 8;
    #pragma unroll
    for (int j = 0; j < 2; ++j) {
      int col = tn * 32 + j * 16 + l15;
      #pragma unroll
      for (int r = 0; r < 8; ++r)
        out[(size_t)(tok + r) * kDim + col] = acc[i][j][r];
    }
  }
}

// ---------------------------------------------------------------------------
extern "C" void kernel_launch(void* const* d_in, const int* in_sizes, int n_in,
                              void* d_out, int out_size, void* d_ws, size_t ws_size,
                              hipStream_t stream) {
  const float* x           = (const float*)d_in[0];
  const float* gamma       = (const float*)d_in[1];
  const float* w_qkv       = (const float*)d_in[2];
  const float* temperature = (const float*)d_in[3];
  const float* w_out       = (const float*)d_in[4];
  float* out = (float*)d_out;

  char* p = (char*)d_ws;
  auto alloc = [&](size_t bytes) {
    char* r = p;
    p += (bytes + 255) & ~size_t(255);
    return r;
  };
  bf16_t* xn    = (bf16_t*)alloc((size_t)kToks * kDim * 2);   // 32 MB
  bf16_t* wqkvT = (bf16_t*)alloc((size_t)kQkvF * kDim * 2);   // 1.5 MB
  bf16_t* woutT = (bf16_t*)alloc((size_t)kDim * kDim * 2);    // 0.5 MB
  bf16_t* qkvT  = (bf16_t*)alloc((size_t)kQkvF * kSeq * 2);   // 96 MB (L2-resident)
  float*  invn  = (float*) alloc((size_t)8 * 1024 * 4);       // 32 KB
  bf16_t* attn  = (bf16_t*)alloc((size_t)64 * 64 * 64 * 2);   // 0.5 MB
  bf16_t* AO    = (bf16_t*)alloc((size_t)kToks * kDim * 2);   // 32 MB

  prep_weights_kernel<<<4096, 256, 0, stream>>>(w_qkv, w_out, wqkvT, woutT);
  rmsnorm_kernel     <<<4096, 256, 0, stream>>>(x, gamma, xn);
  qkv_gemm_kernel    <<<3072, 256, 0, stream>>>(wqkvT, xn, qkvT);
  rownorm_kernel     <<<1024, 256, 0, stream>>>(qkvT, invn);
  sim_softmax_kernel <<<64, 256, 0, stream>>>(qkvT, invn, temperature, attn);
  attn_v_kernel      <<<4096, 256, 0, stream>>>(qkvT, attn, AO);
  out_gemm_kernel    <<<1024, 256, 0, stream>>>(AO, woutT, out);
}
